// AoA_Decoder_80582176407916
// MI455X (gfx1250) — compile-verified, matrix operand order checked by reading, not run on hardware
//
#include <hip/hip_runtime.h>

// ---------------------------------------------------------------------------
// AoA caption decoder for gfx1250 (MI455X).
//   * bf16 WMMA (v_wmma_f32_16x16x32_bf16, f32 accum) for every GEMM
//   * workgroup-cooperative GEMM: 4 waves cover 128 rows, the shared 64x32
//     weight chunk is double-buffered in LDS via GLOBAL_LOAD_ASYNC_TO_LDS_B128
//     (ASYNCcnt pipeline) -> 4x less L2 weight traffic than per-wave streaming
//   * f32 VALU for softmax / LSTM / GLU; K/V projections cached in bf16
// ---------------------------------------------------------------------------

typedef __attribute__((ext_vector_type(16))) __bf16 v16bf;
typedef __attribute__((ext_vector_type(8)))  float  v8f;
typedef int v4i32 __attribute__((vector_size(16)));

#define DEV __device__ __forceinline__
#define AS_GLOBAL __attribute__((address_space(1)))
#define AS_SHARED __attribute__((address_space(3)))

constexpr int Bb    = 128;          // batch
constexpr int Nimg  = 196;          // image features
constexpr int Dm    = 1024;         // d_model
constexpr int Em    = 1024;         // embed dim
constexpr int Vocab = 10000;
constexpr int NHh   = 8;
constexpr int Tmax  = 20;
constexpr int DHh   = 128;          // Dm / NH
constexpr int Kcat  = 3072;         // E + D + D fused LSTM input
constexpr int Ngates= 4096;         // 4*D
constexpr int Vpad  = 10048;        // 157*64, padded vocab for tiling
constexpr int MN    = Bb * Nimg;    // 25088 rows for K/V projection

union FragAB { v16bf v; uint4 u[2]; };

DEV float sigmoidf(float x) { return 1.0f / (1.0f + expf(-x)); }

// --- async global->LDS copy (16B per lane), guarded with safe fallback -----
#if defined(__has_builtin)
#if __has_builtin(__builtin_amdgcn_global_load_async_to_lds_b128) && \
    __has_builtin(__builtin_amdgcn_s_wait_asynccnt)
#define HAS_ASYNC_LDS 1
#endif
#endif
#ifndef HAS_ASYNC_LDS
#define HAS_ASYNC_LDS 0
#endif

DEV void async_copy16(const __bf16* g, __bf16* l)
{
#if HAS_ASYNC_LDS
    __builtin_amdgcn_global_load_async_to_lds_b128(
        (AS_GLOBAL v4i32*)(void*)g,
        (AS_SHARED v4i32*)(void*)l, 0, 0);
#else
    *(uint4*)l = *(const uint4*)g;
#endif
}

DEV void async_wait_2() {
#if HAS_ASYNC_LDS
    __builtin_amdgcn_s_wait_asynccnt(2);
#endif
}
DEV void async_wait_0() {
#if HAS_ASYNC_LDS
    __builtin_amdgcn_s_wait_asynccnt(0);
#endif
}

// ---------------------------------------------------------------------------
// Workgroup GEMM:  C[M,N] = A[M,K] * W[N,K]^T + bias   (W in PyTorch layout)
// Block = 128 threads (4 waves). Workgroup tile = 128(M) x 64(N); wave w owns
// rows [32w,32w+32). The 64x32 W chunk per K-step is staged to LDS once
// (async, double buffered, 80B row pitch -> conflict-free ds_load_b128) and
// consumed by all 4 waves: 2 A-frags x 4 B-frags -> 8 WMMAs per 32-K step.
// A 16-bit fragment (ISA 7.12.2): lane<16 holds row, K chunks {k..k+7} and
// {k+16..k+23}; lane>=16 holds {k+8..k+15} / {k+24..k+31}.
// PRED=true adds the predict-layer epilogue (length mask, V tail, [B,T,V]).
// ---------------------------------------------------------------------------
constexpr int PITCH = 40;   // bf16 elems per LDS row (80 B = 5x16B, banks spread)

template <bool PRED>
__global__ __launch_bounds__(128)
void k_gemm_ws(const __bf16* __restrict__ A, const __bf16* __restrict__ W,
               const float* __restrict__ bias, void* __restrict__ Cout,
               int out_bf16, const int* __restrict__ lengths, int t,
               int N, int K)
{
    __shared__ __bf16 sw[2][64 * PITCH];

    const int tid  = threadIdx.x;
    const int lane = tid & 31;
    const int wave = tid >> 5;
    const int row  = lane & 15;
    const int hi   = lane >> 4;
    const int koff = hi * 8;

    const long n0 = (long)blockIdx.x * 64;
    const long m0 = (long)blockIdx.y * 128 + wave * 32;

    const __bf16* a0 = A + (m0 + row) * (long)K;
    const __bf16* a1 = a0 + 16L * K;

    // staging: 256 16-byte chunks per K-step (64 rows x 4 quarters),
    // 2 chunks per thread.
    const int r0 = tid >> 2,          q0 = tid & 3;
    const int r1 = (tid + 128) >> 2,  q1 = tid & 3;   // (tid+128)&3 == tid&3
    const __bf16* wg0 = W + (n0 + r0) * (long)K + q0 * 8;
    const __bf16* wg1 = W + (n0 + r1) * (long)K + q1 * 8;
    const int l0 = r0 * PITCH + q0 * 8;
    const int l1 = r1 * PITCH + q1 * 8;

    v8f acc[2][4];
#pragma unroll
    for (int i = 0; i < 2; ++i)
#pragma unroll
        for (int j = 0; j < 4; ++j)
#pragma unroll
            for (int r = 0; r < 8; ++r) acc[i][j][r] = 0.0f;

    const int nk = K / 32;
    async_copy16(wg0, &sw[0][l0]);
    async_copy16(wg1, &sw[0][l1]);

    for (int ks = 0; ks < nk; ++ks) {
        if (ks + 1 < nk) {
            const int kn = (ks + 1) * 32;
            const int p  = (ks + 1) & 1;
            async_copy16(wg0 + kn, &sw[p][l0]);
            async_copy16(wg1 + kn, &sw[p][l1]);
            async_wait_2();            // current buffer's 2 loads complete
        } else {
            async_wait_0();
        }
        __syncthreads();               // all waves' staging visible

        const __bf16* sb = sw[ks & 1];
        const int k = ks * 32;

        FragAB fa[2], fw[4];
        fa[0].u[0] = *(const uint4*)(a0 + k + koff);
        fa[0].u[1] = *(const uint4*)(a0 + k + 16 + koff);
        fa[1].u[0] = *(const uint4*)(a1 + k + koff);
        fa[1].u[1] = *(const uint4*)(a1 + k + 16 + koff);
#pragma unroll
        for (int j = 0; j < 4; ++j) {
            const __bf16* wr = sb + (j * 16 + row) * PITCH;
            fw[j].u[0] = *(const uint4*)(wr + koff);
            fw[j].u[1] = *(const uint4*)(wr + 16 + koff);
        }
#pragma unroll
        for (int i = 0; i < 2; ++i)
#pragma unroll
            for (int j = 0; j < 4; ++j)
                acc[i][j] = __builtin_amdgcn_wmma_f32_16x16x32_bf16(
                    false, fa[i].v, false, fw[j].v, (short)0, acc[i][j],
                    false, false);

        __syncthreads();               // protect buffer reuse two steps out
    }

    // C layout: lane gives N = lane&15; VGPR r gives M = r + 8*hi.
#pragma unroll
    for (int j = 0; j < 4; ++j) {
        const long n = n0 + j * 16 + row;
        if (PRED) {
            if (n >= Vocab) continue;
            const float bv = bias[n];
#pragma unroll
            for (int i = 0; i < 2; ++i)
#pragma unroll
                for (int r = 0; r < 8; ++r) {
                    const long b = m0 + i * 16 + hi * 8 + r;   // batch row
                    const bool act = (t < lengths[b]);
                    ((float*)Cout)[((long)b * Tmax + t) * Vocab + n] =
                        act ? (acc[i][j][r] + bv) : 0.0f;
                }
        } else {
            const float bv = bias ? bias[n] : 0.0f;
#pragma unroll
            for (int i = 0; i < 2; ++i)
#pragma unroll
                for (int r = 0; r < 8; ++r) {
                    const long m = m0 + i * 16 + hi * 8 + r;
                    const float val = acc[i][j][r] + bv;
                    if (out_bf16) ((__bf16*)Cout)[m * N + n] = (__bf16)val;
                    else          ((float*)Cout)[m * N + n]  = val;
                }
        }
    }
}

// ---------------------------------------------------------------------------
// Elementwise / setup kernels
// ---------------------------------------------------------------------------
__global__ void k_f32_to_bf16(const float* __restrict__ in,
                              __bf16* __restrict__ out, long n)
{
    long i = (long)blockIdx.x * blockDim.x + threadIdx.x;
    if (i < n) out[i] = (__bf16)in[i];
}

// Fuse W_ih [4096,2048] and W_hh [4096,1024] along K into [4096,3072] bf16.
__global__ void k_build_wcat(const float* __restrict__ W_ih,
                             const float* __restrict__ W_hh,
                             __bf16* __restrict__ Wcat)
{
    long i = (long)blockIdx.x * blockDim.x + threadIdx.x;
    if (i >= (long)Ngates * Kcat) return;
    long r = i / Kcat, c = i % Kcat;
    float v = (c < 2048) ? W_ih[r * 2048 + c] : W_hh[r * 1024 + (c - 2048)];
    Wcat[i] = (__bf16)v;
}

__global__ void k_bias_sum(const float* a, const float* b, float* c, int n)
{
    int i = blockIdx.x * blockDim.x + threadIdx.x;
    if (i < n) c[i] = a[i] + b[i];
}

// Weight-normed predict matrix: Wn[v,:] = g[v]*pred_v[v,:]/||pred_v[v,:]||.
__global__ __launch_bounds__(256)
void k_weightnorm(const float* __restrict__ pv, const float* __restrict__ pg,
                  __bf16* __restrict__ Wn)
{
    const int v = blockIdx.x;                    // 0..Vpad-1
    const int tid = threadIdx.x;
    __shared__ float red[256];
    if (v >= Vocab) {                            // zero the padded tail rows
        for (int d = tid; d < Dm; d += 256) Wn[(long)v * Dm + d] = (__bf16)0.0f;
        return;
    }
    const float* rowp = pv + (long)v * Dm;
    float s = 0.0f;
    for (int d = tid; d < Dm; d += 256) { float x = rowp[d]; s += x * x; }
    red[tid] = s; __syncthreads();
    for (int w = 128; w > 0; w >>= 1) {
        if (tid < w) red[tid] += red[tid + w];
        __syncthreads();
    }
    const float inv = pg[v] * rsqrtf(red[0]);
    for (int d = tid; d < Dm; d += 256)
        Wn[(long)v * Dm + d] = (__bf16)(rowp[d] * inv);
}

__global__ void k_mean_features(const float* __restrict__ enc,
                                float* __restrict__ mean_f)
{
    int i = blockIdx.x * blockDim.x + threadIdx.x;     // B*D threads
    if (i >= Bb * Dm) return;
    int b = i >> 10, d = i & (Dm - 1);
    float s = 0.0f;
    const float* p = enc + ((long)b * Nimg) * Dm + d;
    for (int n = 0; n < Nimg; ++n) s += p[(long)n * Dm];
    mean_f[i] = s * (1.0f / (float)Nimg);
}

__global__ void k_init_state(float* m, float* ctx, __bf16* hb)
{
    int i = blockIdx.x * blockDim.x + threadIdx.x;
    if (i >= Bb * Dm) return;
    m[i] = 0.0f; ctx[i] = 0.0f; hb[i] = (__bf16)0.0f;
}

// x_cat = [ embed(tok) | mean_f + ctx | h ]  -> bf16 [B, 3072]
__global__ void k_build_x(const int* __restrict__ captions,
                          const float* __restrict__ embed_W,
                          const float* __restrict__ mean_f,
                          const float* __restrict__ ctx,
                          const __bf16* __restrict__ hb,
                          __bf16* __restrict__ xb, int t)
{
    long i = (long)blockIdx.x * blockDim.x + threadIdx.x;
    if (i >= (long)Bb * Kcat) return;
    int b = (int)(i / Kcat), c = (int)(i % Kcat);
    __bf16 v;
    if (c < Em) {
        int tok = captions[b * Tmax + t];
        v = (__bf16)embed_W[(long)tok * Em + c];
    } else if (c < Em + Dm) {
        int d = c - Em;
        v = (__bf16)(mean_f[b * Dm + d] + ctx[b * Dm + d]);
    } else {
        v = hb[b * Dm + (c - Em - Dm)];
    }
    xb[i] = v;
}

// LSTM cell elementwise (gates order i,f,g,o). Writes h as bf16 into both the
// h-state buffer and the second half of the AoA input concat.
__global__ void k_lstm(const float* __restrict__ gates, float* __restrict__ m,
                       __bf16* __restrict__ hb, __bf16* __restrict__ aoa_in)
{
    int i = blockIdx.x * blockDim.x + threadIdx.x;
    if (i >= Bb * Dm) return;
    int b = i >> 10, d = i & (Dm - 1);
    const float* g = gates + (long)b * Ngates;
    float ig = sigmoidf(g[d]);
    float fg = sigmoidf(g[Dm + d]);
    float gg = tanhf(g[2 * Dm + d]);
    float og = sigmoidf(g[3 * Dm + d]);
    float m2 = fg * m[i] + ig * gg;
    m[i] = m2;
    float h2 = og * tanhf(m2);
    __bf16 hbv = (__bf16)h2;
    hb[i] = hbv;
    aoa_in[(long)b * (2 * Dm) + Dm + d] = hbv;
}

// Single-query multi-head attention. One workgroup per (batch, head).
__global__ __launch_bounds__(256)
void k_attention(const float* __restrict__ q, const __bf16* __restrict__ Kp,
                 const __bf16* __restrict__ Vp, __bf16* __restrict__ aoa_in)
{
    const int b = blockIdx.x, h = blockIdx.y, tid = threadIdx.x;
    __shared__ float sc[Nimg];
    __shared__ float red[256];
    const float scale = 0.08838834764831845f;     // 1/sqrt(128)
    const float* qv = q + (long)b * Dm + h * DHh;

    if (tid < Nimg) {
        const __bf16* kr = Kp + ((long)b * Nimg + tid) * Dm + h * DHh;
        float s = 0.0f;
        for (int d = 0; d < DHh; ++d) s += qv[d] * (float)kr[d];
        sc[tid] = s * scale;
    }
    __syncthreads();

    float v = (tid < Nimg) ? sc[tid] : -3.4e38f;
    red[tid] = v; __syncthreads();
    for (int w = 128; w > 0; w >>= 1) {
        if (tid < w) red[tid] = fmaxf(red[tid], red[tid + w]);
        __syncthreads();
    }
    const float mx = red[0];
    __syncthreads();

    float e = (tid < Nimg) ? expf(sc[tid] - mx) : 0.0f;
    red[tid] = e; __syncthreads();
    for (int w = 128; w > 0; w >>= 1) {
        if (tid < w) red[tid] += red[tid + w];
        __syncthreads();
    }
    const float denom = red[0];
    __syncthreads();
    if (tid < Nimg) sc[tid] = e / denom;
    __syncthreads();

    if (tid < DHh) {
        const __bf16* vr = Vp + ((long)b * Nimg) * Dm + h * DHh + tid;
        float a = 0.0f;
        for (int n = 0; n < Nimg; ++n) a += sc[n] * (float)vr[(long)n * Dm];
        aoa_in[(long)b * (2 * Dm) + h * DHh + tid] = (__bf16)a;
    }
}

// GLU: ctx2 = a1 * sigmoid(a2); write f32 state + bf16 copy for predict GEMM.
__global__ void k_glu(const float* __restrict__ a, float* __restrict__ ctx,
                      __bf16* __restrict__ ctxb)
{
    int i = blockIdx.x * blockDim.x + threadIdx.x;
    if (i >= Bb * Dm) return;
    int b = i >> 10, d = i & (Dm - 1);
    const float* ar = a + (long)b * (2 * Dm);
    float c = ar[d] * sigmoidf(ar[Dm + d]);
    ctx[i] = c;
    ctxb[i] = (__bf16)c;
}

// ---------------------------------------------------------------------------
// Host-side orchestration
// ---------------------------------------------------------------------------
extern "C" void kernel_launch(void* const* d_in, const int* in_sizes, int n_in,
                              void* d_out, int out_size, void* d_ws, size_t ws_size,
                              hipStream_t stream)
{
    (void)in_sizes; (void)n_in; (void)out_size; (void)ws_size;

    const float* enc      = (const float*)d_in[0];
    const int*   captions = (const int*)  d_in[1];
    const int*   lengths  = (const int*)  d_in[2];
    const float* embed_W  = (const float*)d_in[3];
    const float* W_ih     = (const float*)d_in[4];
    const float* b_ih     = (const float*)d_in[5];
    const float* W_hh     = (const float*)d_in[6];
    const float* b_hh     = (const float*)d_in[7];
    const float* Wq       = (const float*)d_in[8];
    const float* bq       = (const float*)d_in[9];
    const float* Wk       = (const float*)d_in[10];
    const float* bk       = (const float*)d_in[11];
    const float* Wv       = (const float*)d_in[12];
    const float* bv       = (const float*)d_in[13];
    const float* W_aoa    = (const float*)d_in[14];
    const float* b_aoa    = (const float*)d_in[15];
    const float* pred_v   = (const float*)d_in[16];
    const float* pred_g   = (const float*)d_in[17];
    const float* pred_b   = (const float*)d_in[18];
    float* out = (float*)d_out;

    char* ws = (char*)d_ws;
    size_t off = 0;
    auto alloc = [&](size_t bytes) -> void* {
        void* p = ws + off;
        off = (off + bytes + 255) & ~(size_t)255;
        return p;
    };

    __bf16* enc_bf  = (__bf16*)alloc((size_t)MN * Dm * 2);
    __bf16* Kp      = (__bf16*)alloc((size_t)MN * Dm * 2);
    __bf16* Vp      = (__bf16*)alloc((size_t)MN * Dm * 2);
    __bf16* Wk_bf   = (__bf16*)alloc((size_t)Dm * Dm * 2);
    __bf16* Wv_bf   = (__bf16*)alloc((size_t)Dm * Dm * 2);
    __bf16* Wq_bf   = (__bf16*)alloc((size_t)Dm * Dm * 2);
    __bf16* Waoa_bf = (__bf16*)alloc((size_t)2 * Dm * 2 * Dm * 2);
    __bf16* Wcat    = (__bf16*)alloc((size_t)Ngates * Kcat * 2);
    float*  bcat    = (float*) alloc((size_t)Ngates * 4);
    __bf16* Wn      = (__bf16*)alloc((size_t)Vpad * Dm * 2);
    float*  mean_f  = (float*) alloc((size_t)Bb * Dm * 4);
    __bf16* xb      = (__bf16*)alloc((size_t)Bb * Kcat * 2);
    float*  gates   = (float*) alloc((size_t)Bb * Ngates * 4);
    float*  mbuf    = (float*) alloc((size_t)Bb * Dm * 4);
    float*  ctxbuf  = (float*) alloc((size_t)Bb * Dm * 4);
    __bf16* hb      = (__bf16*)alloc((size_t)Bb * Dm * 2);
    float*  qbuf    = (float*) alloc((size_t)Bb * Dm * 4);
    __bf16* aoa_in  = (__bf16*)alloc((size_t)Bb * 2 * Dm * 2);
    float*  aoa_out = (float*) alloc((size_t)Bb * 2 * Dm * 4);
    __bf16* ctxb    = (__bf16*)alloc((size_t)Bb * Dm * 2);

    const int TPB = 256;
    auto blocks = [](long n, int t) { return (unsigned)((n + t - 1) / t); };

    // ---- one-time setup ----
    k_f32_to_bf16<<<blocks((long)MN * Dm, TPB), TPB, 0, stream>>>(enc, enc_bf, (long)MN * Dm);
    k_f32_to_bf16<<<blocks((long)Dm * Dm, TPB), TPB, 0, stream>>>(Wk, Wk_bf, (long)Dm * Dm);
    k_f32_to_bf16<<<blocks((long)Dm * Dm, TPB), TPB, 0, stream>>>(Wv, Wv_bf, (long)Dm * Dm);
    k_f32_to_bf16<<<blocks((long)Dm * Dm, TPB), TPB, 0, stream>>>(Wq, Wq_bf, (long)Dm * Dm);
    k_f32_to_bf16<<<blocks((long)4 * Dm * Dm, TPB), TPB, 0, stream>>>(W_aoa, Waoa_bf, (long)4 * Dm * Dm);
    k_build_wcat<<<blocks((long)Ngates * Kcat, TPB), TPB, 0, stream>>>(W_ih, W_hh, Wcat);
    k_bias_sum<<<blocks(Ngates, TPB), TPB, 0, stream>>>(b_ih, b_hh, bcat, Ngates);
    k_weightnorm<<<Vpad, 256, 0, stream>>>(pred_v, pred_g, Wn);
    k_mean_features<<<blocks((long)Bb * Dm, TPB), TPB, 0, stream>>>(enc, mean_f);
    k_init_state<<<blocks((long)Bb * Dm, TPB), TPB, 0, stream>>>(mbuf, ctxbuf, hb);

    // K/V projections (step-invariant), stored bf16 row-major [B*N, D].
    k_gemm_ws<false><<<dim3(Dm / 64, MN / 128), 128, 0, stream>>>(
        enc_bf, Wk_bf, bk, (void*)Kp, 1, nullptr, 0, Dm, Dm);
    k_gemm_ws<false><<<dim3(Dm / 64, MN / 128), 128, 0, stream>>>(
        enc_bf, Wv_bf, bv, (void*)Vp, 1, nullptr, 0, Dm, Dm);

    // ---- sequential decode over T steps ----
    for (int t = 0; t < Tmax; ++t) {
        k_build_x<<<blocks((long)Bb * Kcat, TPB), TPB, 0, stream>>>(
            captions, embed_W, mean_f, ctxbuf, hb, xb, t);
        // fused LSTM gates: [128,3072] x [3072,4096]
        k_gemm_ws<false><<<dim3(Ngates / 64, 1), 128, 0, stream>>>(
            xb, Wcat, bcat, (void*)gates, 0, nullptr, 0, Ngates, Kcat);
        k_lstm<<<blocks((long)Bb * Dm, TPB), TPB, 0, stream>>>(gates, mbuf, hb, aoa_in);
        // q projection: [128,1024] x [1024,1024]
        k_gemm_ws<false><<<dim3(Dm / 64, 1), 128, 0, stream>>>(
            hb, Wq_bf, bq, (void*)qbuf, 0, nullptr, 0, Dm, Dm);
        k_attention<<<dim3(Bb, NHh), 256, 0, stream>>>(qbuf, Kp, Vp, aoa_in);
        // AoA: [128,2048] x [2048,2048]
        k_gemm_ws<false><<<dim3(2 * Dm / 64, 1), 128, 0, stream>>>(
            aoa_in, Waoa_bf, b_aoa, (void*)aoa_out, 0, nullptr, 0, 2 * Dm, 2 * Dm);
        k_glu<<<blocks((long)Bb * Dm, TPB), TPB, 0, stream>>>(aoa_out, ctxbuf, ctxb);
        // predict: [128,1024] x [1024,10048pad] -> [B,T,V] with length mask
        k_gemm_ws<true><<<dim3(Vpad / 64, 1), 128, 0, stream>>>(
            ctxb, Wn, pred_b, (void*)out, 0, lengths, t, Vocab, Dm);
    }
}
